// DynANet_54099408060629
// MI455X (gfx1250) — compile-verified, matrix operand order checked by reading, not run on hardware
//
#include <hip/hip_runtime.h>
#include <cstdint>
#include <cstddef>

typedef _Float16 half_t;
typedef __attribute__((ext_vector_type(16))) _Float16 v16h;
typedef __attribute__((ext_vector_type(8)))  float    v8f;

static constexpr int B    = 8;
static constexpr int N    = 2048;
static constexpr int KNN  = 16;
static constexpr int S    = 500;
static constexpr int NPTS = B * N;        // 16384
static constexpr int CAT  = 672;          // 32 + 128 + 512
static constexpr int GEW  = 128;
static constexpr int GFW  = 832;          // 803 padded to mult of 32
static constexpr int NROWS = B * S;       // 4000

#define DEV __device__ __forceinline__

DEV v8f zero8() {
  v8f c;
#pragma unroll
  for (int i = 0; i < 8; ++i) c[i] = 0.0f;
  return c;
}

// Load a 16-half WMMA operand slice for this lane.
// A-operand (16x32 f16): lane<16 -> row M=lane, halves {kb..kb+7, kb+16..kb+23} with kb=kc*32;
// lane>=16 -> same row set M=lane-16 but kb offset +8.  B-operand (32x16) is the symmetric
// per-column layout, so the same loader works with "row" = weight row n of W^T.
DEV v16h load_ab(const half_t* p, int kb) {
  v16h a;
#pragma unroll
  for (int i = 0; i < 8; ++i) { a[i] = p[kb + i]; a[i + 8] = p[kb + 16 + i]; }
  return a;
}

template <int KP>
DEV void gemm_tile(const half_t* arow, const half_t* brow, int off, v8f& c) {
#pragma unroll
  for (int kc = 0; kc < KP / 32; ++kc) {
    v16h a = load_ab(arow, kc * 32 + off);
    v16h b = load_ab(brow, kc * 32 + off);
    c = __builtin_amdgcn_wmma_f32_16x16x32_f16(false, a, false, b, (short)0, c,
                                               false, false);
  }
}

// One 16xKP @ KPxH WMMA layer: A rows (stride astride, f16), W^T [H][KP] f16, bias f32[H].
// Output written f16 to obase (stride ostride) with optional ReLU.  Single-wave block.
template <int KP, int H>
DEV void layer16(const half_t* abase, int astride, const half_t* wt,
                 const float* bias, half_t* obase, int ostride, bool relu,
                 int lane) {
  const int col = lane & 15;
  const int off = (lane >= 16) ? 8 : 0;
  const half_t* arow = abase + (size_t)col * astride;
#pragma unroll
  for (int nt = 0; nt < H / 16; ++nt) {
    const half_t* brow = wt + (size_t)(nt * 16 + col) * KP;
    v8f c = zero8();
    gemm_tile<KP>(arow, brow, off, c);
    float bv = bias[nt * 16 + col];
#pragma unroll
    for (int r = 0; r < 8; ++r) {
      float v = c[r] + bv;
      if (relu) v = fmaxf(v, 0.0f);
      obase[(size_t)(r + off) * ostride + nt * 16 + col] = (half_t)v;
    }
  }
  __syncthreads();
}

// Same, but final layer: write f32 to global [row][H] at rowbase.
template <int KP, int H>
DEV void layer16_glob(const half_t* abase, int astride, const half_t* wt,
                      const float* bias, float* gout, int gstride,
                      size_t rowbase, int lane) {
  const int col = lane & 15;
  const int off = (lane >= 16) ? 8 : 0;
  const half_t* arow = abase + (size_t)col * astride;
#pragma unroll
  for (int nt = 0; nt < H / 16; ++nt) {
    const half_t* brow = wt + (size_t)(nt * 16 + col) * KP;
    v8f c = zero8();
    gemm_tile<KP>(arow, brow, off, c);
    float bv = bias[nt * 16 + col];
#pragma unroll
    for (int r = 0; r < 8; ++r)
      gout[(rowbase + r + off) * (size_t)gstride + nt * 16 + col] = c[r] + bv;
  }
  __syncthreads();
}

// ---------------- kNN: fused WMMA Gram + register top-16 ----------------
// One wave per 16-query tile.  Distances d = |xi|^2 + |xj|^2 - 2*Gram via
// v_wmma_f32_16x16x32_f16; each lane keeps an ascending top-16 of half the
// candidates of its query; pairs of lanes bitonic-merge at the end.
template <int FP>
__global__ void __launch_bounds__(32)
knn_kernel(const half_t* __restrict__ x, const float* __restrict__ x2,
           int* __restrict__ nbr) {
  __shared__ float tile[16 * 17];
  const int lane  = threadIdx.x;
  const int ntile = N / 16;
  const int bb    = blockIdx.x / ntile;
  const int qbase = (blockIdx.x % ntile) * 16;
  const int col   = lane & 15;
  const int off   = (lane >= 16) ? 8 : 0;

  v16h A[FP / 32];
  const half_t* qrow = x + ((size_t)bb * N + qbase + col) * FP;
#pragma unroll
  for (int kc = 0; kc < FP / 32; ++kc) A[kc] = load_ab(qrow, kc * 32 + off);

  float xq2[8];
#pragma unroll
  for (int r = 0; r < 8; ++r) xq2[r] = x2[bb * N + qbase + off + r];

  float dist[16]; int idx[16];
#pragma unroll
  for (int i = 0; i < 16; ++i) { dist[i] = 1e30f; idx[i] = 0; }

  for (int ct = 0; ct < ntile; ++ct) {
    const int cbase = ct * 16;
    const half_t* crow = x + ((size_t)bb * N + cbase + col) * FP;
    v8f c = zero8();
#pragma unroll
    for (int kc = 0; kc < FP / 32; ++kc) {
      v16h Bv = load_ab(crow, kc * 32 + off);
      c = __builtin_amdgcn_wmma_f32_16x16x32_f16(false, A[kc], false, Bv,
                                                 (short)0, c, false, false);
    }
    const float xc2 = x2[bb * N + cbase + col];
#pragma unroll
    for (int r = 0; r < 8; ++r)
      tile[(r + off) * 17 + col] = xq2[r] + xc2 - 2.0f * c[r];
    __syncthreads();
#pragma unroll
    for (int i = 0; i < 8; ++i) {
      float dv = tile[col * 17 + off + i];
      int   j  = cbase + off + i;
      if (dv < dist[15]) {                 // replace max, bubble down (sorted asc)
        dist[15] = dv; idx[15] = j;
#pragma unroll
        for (int t = 15; t > 0; --t) {
          float d0 = dist[t - 1], d1 = dist[t];
          bool  sw = d1 < d0;
          dist[t - 1] = sw ? d1 : d0;  dist[t] = sw ? d0 : d1;
          int i0 = idx[t - 1], i1 = idx[t];
          idx[t - 1] = sw ? i1 : i0;   idx[t]  = sw ? i0 : i1;
        }
      }
    }
    __syncthreads();
  }
  // merge lane pair (l, l+16): both ascending; min(A[i], B[15-i]) = 16 smallest of union
  int outi[16];
#pragma unroll
  for (int i = 0; i < 16; ++i) {
    float pd = __shfl_xor(dist[15 - i], 16, 32);
    int   pi = __shfl_xor(idx[15 - i], 16, 32);
    outi[i]  = (dist[i] <= pd) ? idx[i] : pi;
  }
  if (lane < 16) {
#pragma unroll
    for (int i = 0; i < 16; ++i)
      nbr[((size_t)bb * N + qbase + col) * KNN + i] = outi[i];
  }
}

// ---------------- EdgeConv: 16 edges of one point per wave ----------------
template <int FP, int H1, int H2, int H3>
__global__ void __launch_bounds__(32)
edge_conv_kernel(const half_t* __restrict__ x, const int* __restrict__ nbr,
                 const half_t* w1, const float* b1, const half_t* w2,
                 const float* b2, const half_t* w3, const float* b3,
                 float* __restrict__ xcat, int cat_off,
                 half_t* __restrict__ xnext) {
  constexpr int K1 = 2 * FP;
  constexpr int K2 = (H1 < 32) ? 32 : H1;
  constexpr int K3 = (H2 < 32) ? 32 : H2;
  constexpr int BUF0 = (K1 > H3) ? K1 : H3;
  constexpr int BUF1 = (BUF0 > K2) ? BUF0 : K2;
  constexpr int BUF  = (BUF1 > K3) ? BUF1 : K3;
  __shared__ half_t bufA[16 * BUF];
  __shared__ half_t bufB[16 * BUF];

  const int lane = threadIdx.x;
  const int q    = blockIdx.x;           // global point index
  const int bb   = q / N;

  for (int i = lane; i < 16 * BUF; i += 32) {
    bufA[i] = (half_t)0.0f; bufB[i] = (half_t)0.0f;
  }
  __syncthreads();

  { // build edge features [x_i | x_j - x_i], zero-padded layout
    const int r = lane & 15, seg = lane >> 4;
    const int j = nbr[q * KNN + r];
    const half_t* xi = x + (size_t)q * FP;
    const half_t* xj = x + ((size_t)bb * N + j) * FP;
    half_t* row = bufA + r * BUF + seg * FP;
    for (int c = 0; c < FP; ++c) {
      float v = seg ? ((float)xj[c] - (float)xi[c]) : (float)xi[c];
      row[c] = (half_t)v;
    }
  }
  __syncthreads();

  layer16<K1, H1>(bufA, BUF, w1, b1, bufB, BUF, true,  lane);
  layer16<K2, H2>(bufB, BUF, w2, b2, bufA, BUF, true,  lane);
  layer16<K3, H3>(bufA, BUF, w3, b3, bufB, BUF, false, lane);

  for (int c = lane; c < H3; c += 32) {   // max over the 16 edges
    float m = -1e30f;
#pragma unroll 4
    for (int r = 0; r < 16; ++r) m = fmaxf(m, (float)bufB[r * BUF + c]);
    xcat[(size_t)q * CAT + cat_off + c] = m;
    if (xnext) xnext[(size_t)q * H3 + c] = (half_t)m;
  }
}

// ---------------- shared / grasp MLPs ----------------
__global__ void __launch_bounds__(32)
shared_mlp_kernel(const half_t* __restrict__ xcat16, const half_t* w1,
                  const float* b1, const half_t* w2, const float* b2,
                  float* __restrict__ sh) {
  __shared__ half_t buf[16 * 256];
  const int lane = threadIdx.x;
  const size_t ptbase = (size_t)blockIdx.x * 16;
  layer16<CAT, 256>(xcat16 + ptbase * CAT, CAT, w1, b1, buf, 256, true, lane);
  layer16_glob<256, 128>(buf, 256, w2, b2, sh, 128, ptbase, lane);
}

__global__ void __launch_bounds__(32)
grasp_mlp_kernel(const half_t* __restrict__ gf, const half_t* w1,
                 const float* b1, const half_t* w2, const float* b2,
                 float* __restrict__ out) {
  __shared__ half_t buf[16 * 64];
  const int lane = threadIdx.x;
  const size_t rb = (size_t)blockIdx.x * 16;
  layer16<GFW, 64>(gf + rb * GFW, GFW, w1, b1, buf, 64, true, lane);
  layer16_glob<64, 16>(buf, 64, w2, b2, out, 16, rb, lane);
}

// ---------------- auxiliary kernels ----------------
__global__ void prep_weight_kernel(const float* __restrict__ w,
                                   half_t* __restrict__ wt, int fin, int fout,
                                   int finp, int fsplit) {
  int t = blockIdx.x * blockDim.x + threadIdx.x;
  if (t >= fout * finp) return;
  int n = t / finp, c = t % finp;
  int src = -1;
  if (fsplit > 0) {                       // edge layout: [xi(F)|pad | xj-xi(F)|pad]
    int h2 = finp >> 1;
    if (c < fsplit) src = c;
    else if (c >= h2 && c < h2 + fsplit) src = c - h2 + fsplit;
  } else if (c < fin) {
    src = c;
  }
  wt[t] = (src >= 0) ? (half_t)w[(size_t)src * fout + n] : (half_t)0.0f;
}

__global__ void pos16_kernel(const float* __restrict__ pos,
                             half_t* __restrict__ xp) {
  int p = blockIdx.x * blockDim.x + threadIdx.x;
  if (p >= NPTS) return;
  for (int c = 0; c < 32; ++c)
    xp[(size_t)p * 32 + c] = (c < 3) ? (half_t)pos[p * 3 + c] : (half_t)0.0f;
}

__global__ void sumsq_kernel(const float* __restrict__ src, int stride, int off,
                             int w, float* __restrict__ x2) {
  int p = blockIdx.x * blockDim.x + threadIdx.x;
  if (p >= NPTS) return;
  float s = 0.0f;
  for (int c = 0; c < w; ++c) {
    float v = src[(size_t)p * stride + off + c];
    s += v * v;
  }
  x2[p] = s;
}

__global__ void cvt16_kernel(const float* __restrict__ src,
                             half_t* __restrict__ dst, int n) {
  int t = blockIdx.x * blockDim.x + threadIdx.x;
  if (t < n) dst[t] = (half_t)src[t];
}

__global__ void cls_kernel(const float* __restrict__ sh,
                           const float* __restrict__ wc,
                           const float* __restrict__ bc,
                           float* __restrict__ cls) {
  int p = blockIdx.x * blockDim.x + threadIdx.x;
  if (p >= NPTS) return;
  float a = bc[0];
  for (int c = 0; c < 128; ++c) a += sh[(size_t)p * 128 + c] * wc[c];
  cls[p] = 1.0f / (1.0f + expf(-a));
}

__global__ void gmax_kernel(const float* __restrict__ sh,
                            float* __restrict__ ge) {
  int b = blockIdx.x, c = threadIdx.x;     // 128 threads
  float m = -1e30f;
  for (int n = 0; n < N; ++n)
    m = fmaxf(m, sh[((size_t)b * N + n) * 128 + c]);
  ge[b * 128 + c] = m;
}

__global__ void cdf_kernel(const float* __restrict__ cls,
                           float* __restrict__ cdf) {
  int b = blockIdx.x;
  if (threadIdx.x == 0) {
    float run = 0.0f;
    for (int n = 0; n < N; ++n) { run += cls[b * N + n]; cdf[b * N + n] = run; }
  }
}

__global__ void sample_kernel(const float* __restrict__ cdf,
                              const float* __restrict__ pos,
                              const float* __restrict__ y,
                              int* __restrict__ pidx, float* __restrict__ ap,
                              float* __restrict__ gt) {
  int r = blockIdx.x * blockDim.x + threadIdx.x;
  if (r >= NROWS) return;
  int b = r / S;
  unsigned h = (unsigned)r * 0x9E3779B9u + 0x85EBCA6Bu;
  h ^= h >> 16; h *= 0x7FEB352Du; h ^= h >> 15; h *= 0x846CA68Bu; h ^= h >> 16;
  float u = (h >> 8) * (1.0f / 16777216.0f);
  float t = u * cdf[b * N + N - 1];
  int lo = 0, hi = N - 1;
  while (lo < hi) {
    int mid = (lo + hi) >> 1;
    if (cdf[b * N + mid] < t) lo = mid + 1; else hi = mid;
  }
  int fi = b * N + lo;
  pidx[r] = fi;
  for (int c = 0; c < 3; ++c)  ap[r * 3 + c]  = pos[fi * 3 + c];
  for (int c = 0; c < 16; ++c) gt[r * 16 + c] = y[(size_t)fi * 16 + c];
}

__global__ void gf_kernel(const float* __restrict__ xcat,
                          const float* __restrict__ ge,
                          const float* __restrict__ ap,
                          const int* __restrict__ pidx,
                          half_t* __restrict__ gf) {
  int t = blockIdx.x * blockDim.x + threadIdx.x;
  if (t >= NROWS * GFW) return;
  int r = t / GFW, c = t % GFW;
  float v = 0.0f;
  if (c < CAT)            v = xcat[(size_t)pidx[r] * CAT + c];
  else if (c < CAT + GEW) v = ge[(r % B) * GEW + (c - CAT)];
  else if (c < 803)       v = ap[r * 3 + (c - 800)];
  gf[t] = (half_t)v;
}

// ---------------- host: launch pipeline ----------------
extern "C" void kernel_launch(void* const* d_in, const int* in_sizes, int n_in,
                              void* d_out, int out_size, void* d_ws,
                              size_t ws_size, hipStream_t stream) {
  (void)in_sizes; (void)n_in; (void)out_size; (void)ws_size;
  const float* pos = (const float*)d_in[0];
  const float* y   = (const float*)d_in[2];
  // params flattened in dict order: conv1(w,b x3), conv2, conv3, shared(x2), cls, grasp(x2)
  const float* W[13]; const float* Bb[13];
  {
    int base = 4;
    for (int i = 0; i < 13; ++i) { W[i] = (const float*)d_in[base + 2 * i];
                                   Bb[i] = (const float*)d_in[base + 2 * i + 1]; }
  }
  // weight specs: {fin, fout, finp, fsplit}
  const int spec[13][4] = {
      {6, 16, 64, 3},   {16, 16, 32, 0},  {16, 32, 32, 0},     // conv1
      {64, 64, 64, 32}, {64, 64, 64, 0},  {64, 128, 64, 0},    // conv2
      {256, 256, 256, 128}, {256, 256, 256, 0}, {256, 512, 256, 0}, // conv3
      {672, 256, 672, 0},   {256, 128, 256, 0},                // shared
      {128, 1, 128, 0},                                        // cls (kept f32)
      {803, 64, 832, 0}};                                      // grasp w1
  // grasp w2 handled as spec index 13 below
  const int specG2[4] = {64, 16, 64, 0};

  char* wsp = (char*)d_ws;
  auto alloc = [&](size_t bytes) -> char* {
    char* p = wsp; wsp += (bytes + 255) & ~(size_t)255; return p;
  };
  half_t* wt[14];
  for (int i = 0; i < 13; ++i) {
    if (i == 11) { wt[i] = nullptr; continue; }  // cls stays f32
    wt[i] = (half_t*)alloc((size_t)spec[i][1] * spec[i][2] * sizeof(half_t));
  }
  wt[13] = (half_t*)alloc((size_t)specG2[1] * specG2[2] * sizeof(half_t));

  half_t* xp16   = (half_t*)alloc((size_t)NPTS * 32 * sizeof(half_t));
  half_t* x1_16  = (half_t*)alloc((size_t)NPTS * 32 * sizeof(half_t));
  half_t* x2_16  = (half_t*)alloc((size_t)NPTS * 128 * sizeof(half_t));
  float*  x2sq   = (float*)alloc((size_t)NPTS * sizeof(float));
  int*    nbr    = (int*)alloc((size_t)NPTS * KNN * sizeof(int));
  float*  xcat   = (float*)alloc((size_t)NPTS * CAT * sizeof(float));
  half_t* xcat16 = (half_t*)alloc((size_t)NPTS * CAT * sizeof(half_t));
  float*  sh     = (float*)alloc((size_t)NPTS * 128 * sizeof(float));
  float*  ge     = (float*)alloc((size_t)B * GEW * sizeof(float));
  float*  cdf    = (float*)alloc((size_t)NPTS * sizeof(float));
  int*    pidx   = (int*)alloc((size_t)NROWS * sizeof(int));
  half_t* gf     = (half_t*)alloc((size_t)NROWS * GFW * sizeof(half_t));

  float* out      = (float*)d_out;
  float* out_cls  = out;                       // [16384]
  float* out_gr   = out + NPTS;                // [4000*16]
  float* out_ap   = out_gr + NROWS * 16;       // [4000*3]
  float* out_gt   = out_ap + NROWS * 3;        // [4000*16]

  // -- prep: weights to f16 transposed/padded --
  for (int i = 0; i < 14; ++i) {
    if (i == 11) continue;
    const int* sp = (i == 13) ? specG2 : spec[i];
    const float* ws = (i == 13) ? W[12 + 0 + 0] : W[i];
    // grasp w2 is params index 12? grasp = entries 12(w1),13(w2) in our flat W list:
    // W[11]=cls.w, W[12]=grasp.w1; grasp.w2 lives at d_in[4+2*13]=d_in[30].
    if (i == 13) ws = (const float*)d_in[30];
    int total = sp[1] * sp[2];
    prep_weight_kernel<<<(total + 255) / 256, 256, 0, stream>>>(
        ws, wt[i], sp[0], sp[1], sp[2], sp[3]);
  }
  const float* grasp_b2 = (const float*)d_in[31];

  pos16_kernel<<<(NPTS + 255) / 256, 256, 0, stream>>>(pos, xp16);

  const int knn_blocks = B * (N / 16);

  // stage 1: kNN on pos(3), edge-MLP [6,16,16,32]
  sumsq_kernel<<<(NPTS + 255) / 256, 256, 0, stream>>>(pos, 3, 0, 3, x2sq);
  knn_kernel<32><<<knn_blocks, 32, 0, stream>>>(xp16, x2sq, nbr);
  edge_conv_kernel<32, 16, 16, 32><<<NPTS, 32, 0, stream>>>(
      xp16, nbr, wt[0], Bb[0], wt[1], Bb[1], wt[2], Bb[2], xcat, 0, x1_16);

  // stage 2: kNN on x1(32), edge-MLP [64,64,64,128]
  sumsq_kernel<<<(NPTS + 255) / 256, 256, 0, stream>>>(xcat, CAT, 0, 32, x2sq);
  knn_kernel<32><<<knn_blocks, 32, 0, stream>>>(x1_16, x2sq, nbr);
  edge_conv_kernel<32, 64, 64, 128><<<NPTS, 32, 0, stream>>>(
      x1_16, nbr, wt[3], Bb[3], wt[4], Bb[4], wt[5], Bb[5], xcat, 32, x2_16);

  // stage 3: kNN on x2(128), edge-MLP [256,256,256,512]
  sumsq_kernel<<<(NPTS + 255) / 256, 256, 0, stream>>>(xcat, CAT, 32, 128, x2sq);
  knn_kernel<128><<<knn_blocks, 32, 0, stream>>>(x2_16, x2sq, nbr);
  edge_conv_kernel<128, 256, 256, 512><<<NPTS, 32, 0, stream>>>(
      x2_16, nbr, wt[6], Bb[6], wt[7], Bb[7], wt[8], Bb[8], xcat, 160, nullptr);

  // shared MLP + heads
  cvt16_kernel<<<(NPTS * CAT + 255) / 256, 256, 0, stream>>>(xcat, xcat16,
                                                             NPTS * CAT);
  shared_mlp_kernel<<<NPTS / 16, 32, 0, stream>>>(xcat16, wt[9], Bb[9], wt[10],
                                                  Bb[10], sh);
  cls_kernel<<<(NPTS + 255) / 256, 256, 0, stream>>>(sh, W[11], Bb[11], out_cls);
  gmax_kernel<<<B, 128, 0, stream>>>(sh, ge);

  // sampling + grasp head
  cdf_kernel<<<B, 32, 0, stream>>>(out_cls, cdf);
  sample_kernel<<<(NROWS + 255) / 256, 256, 0, stream>>>(cdf, pos, y, pidx,
                                                         out_ap, out_gt);
  gf_kernel<<<(NROWS * GFW + 255) / 256, 256, 0, stream>>>(xcat, ge, out_ap,
                                                           pidx, gf);
  grasp_mlp_kernel<<<NROWS / 16, 32, 0, stream>>>(gf, wt[12], Bb[12], wt[13],
                                                  grasp_b2, out_gr);
}
